// CCorrM_6442450944214
// MI455X (gfx1250) — matrix operand order repaired and checked by reading, not compiled, not run on hardware
//
#include <hip/hip_runtime.h>
#include <hip/hip_bf16.h>

// ---------------------------------------------------------------------------
// CCorrM co-attention for MI455X (gfx1250, wave32, WMMA + TDM/async-LDS).
//  * A = (Q·Exᵀ)·W_eᵀ refactor kills one C×N GEMM + 134MB intermediate.
//  * All GEMMs on v_wmma_f32_16x16x32_bf16 (8x K-depth vs f32 WMMA), fp32 acc.
//  * BF16 operand staging (c-major + n-major transposed copies) so every WMMA
//    operand is K-contiguous per lane -> 16B vector loads only.
//  * pwconv B-panels DMA'd by the Tensor Data Mover (tensor_load_to_lds,
//    s_wait_tensorcnt); attgemm B-panels via global_load_async_to_lds_b128
//    (s_wait_asynccnt). BN/PReLU/softmax fused into epilogues; fp32 out.
// Workspace requirement: ~406 MB.
// ---------------------------------------------------------------------------

#define B_  16
#define C_  128
#define N_  16384            // H*W
#define BCN ((size_t)B_ * C_ * N_)   // 33,554,432
#define BCC ((size_t)B_ * C_ * C_)   // 262,144

typedef __bf16 bf16;
typedef __attribute__((ext_vector_type(16))) __bf16 v16bf;
typedef __attribute__((ext_vector_type(8)))  __bf16 v8bf;
typedef __attribute__((ext_vector_type(4)))  __bf16 v4bf;
typedef __attribute__((ext_vector_type(8)))  float  v8f;
typedef __attribute__((ext_vector_type(4)))  unsigned u32x4;
typedef __attribute__((ext_vector_type(8)))  unsigned u32x8;

__device__ __forceinline__ bf16 f2bf(float f) {
  unsigned u = __builtin_bit_cast(unsigned, f);
  unsigned r = u + 0x7FFFu + ((u >> 16) & 1u);      // round-to-nearest-even
  unsigned short s = (unsigned short)(r >> 16);
  return __builtin_bit_cast(bf16, s);
}
__device__ __forceinline__ float bf2f(bf16 b) {
  unsigned u = ((unsigned)__builtin_bit_cast(unsigned short, b)) << 16;
  return __builtin_bit_cast(float, u);
}

// ------------------------------ fp32 -> bf16 -------------------------------
__global__ __launch_bounds__(256) void cvt_kernel(const float* __restrict__ src,
                                                  bf16* __restrict__ dst, int n4) {
  int i = blockIdx.x * 256 + threadIdx.x;
  if (i >= n4) return;
  float4 v = *(const float4*)(src + (size_t)i * 4);
  v4bf o;
  o[0] = f2bf(v.x); o[1] = f2bf(v.y); o[2] = f2bf(v.z); o[3] = f2bf(v.w);
  *(v4bf*)(dst + (size_t)i * 4) = o;
}

// --------------------- fp32 [B][C][N] -> bf16 [B][N][C] --------------------
__global__ __launch_bounds__(256) void transpose_kernel(const float* __restrict__ src,
                                                        bf16* __restrict__ dst) {
  __shared__ bf16 tile[32][33];
  int blk = blockIdx.x;                 // B * (C/32) * (N/32) = 16*4*512
  int nt = blk & 511;
  int ct = (blk >> 9) & 3;
  int b  = blk >> 11;
  int tx = threadIdx.x & 31;
  int ty = threadIdx.x >> 5;            // 0..7
  const float* s = src + ((size_t)b * C_ + ct * 32) * N_ + nt * 32;
#pragma unroll
  for (int k = 0; k < 4; k++)
    tile[ty + 8 * k][tx] = f2bf(s[(size_t)(ty + 8 * k) * N_ + tx]);
  __syncthreads();
  bf16* d = dst + ((size_t)b * N_ + nt * 32) * C_ + ct * 32;
#pragma unroll
  for (int k = 0; k < 4; k++)
    d[(size_t)(ty + 8 * k) * C_ + tx] = tile[tx][ty + 8 * k];
}

// ----------------- GEMM1: G[b] = Q[b] @ Ex[b]^T  (K = N = 16384) -----------
__global__ __launch_bounds__(256) void gemm1_kernel(const bf16* __restrict__ qb,
                                                    const bf16* __restrict__ eb,
                                                    float* __restrict__ G) {
  int blk = blockIdx.x;                 // B*8*8 = 1024
  int b  = blk >> 6;
  int mt = (blk >> 3) & 7;
  int nt = blk & 7;
  int wave = threadIdx.x >> 5;
  int lane = threadIdx.x & 31;
  int row = lane & 15;
  int hi  = lane >> 4;
  const bf16* Ar = qb + ((size_t)b * C_ + mt * 16 + row) * N_ + wave * (N_ / 8);
  const bf16* Br = eb + ((size_t)b * C_ + nt * 16 + row) * N_ + wave * (N_ / 8);
  v8f acc = {};
  for (int k = 0; k < N_ / 8; k += 32) {
    __builtin_prefetch(Ar + k + 512, 0, 3);   // near-scope global_prefetch_b8
    __builtin_prefetch(Br + k + 512, 0, 3);
    // A (16x32 bf16): lo lanes K0-7 & 16-23, hi lanes K8-15 & 24-31
    v8bf a0 = *(const v8bf*)(Ar + k + hi * 8);
    v8bf a1 = *(const v8bf*)(Ar + k + 16 + hi * 8);
    // B (32x16 bf16): lane = column, lo lanes K0-7 & 16-23, hi lanes K8-15 & 24-31
    v8bf b0 = *(const v8bf*)(Br + k + hi * 8);
    v8bf b1 = *(const v8bf*)(Br + k + 16 + hi * 8);
    v16bf av, bv;
#pragma unroll
    for (int i = 0; i < 8; i++) {
      av[i] = a0[i]; av[i + 8] = a1[i];
      bv[i] = b0[i]; bv[i + 8] = b1[i];
    }
    acc = __builtin_amdgcn_wmma_f32_16x16x32_bf16(false, av, false, bv,
                                                  (short)0, acc, false, false);
  }
  __shared__ float red[8 * 256];
#pragma unroll
  for (int r = 0; r < 8; r++) red[wave * 256 + r * 32 + lane] = acc[r];
  __syncthreads();
  int t = threadIdx.x;
  float s = 0.f;
#pragma unroll
  for (int w = 0; w < 8; w++) s += red[w * 256 + t];
  int r = t >> 5, l = t & 31;
  int M = r + ((l & 16) ? 8 : 0);
  G[((size_t)b * C_ + mt * 16 + M) * C_ + nt * 16 + (l & 15)] = s;
}

// ----------------- GEMM2: A[b] = G[b] @ W_e^T  (all tiny, K = C) -----------
__global__ __launch_bounds__(256) void gemm2_kernel(const float* __restrict__ G,
                                                    const float* __restrict__ We,
                                                    float* __restrict__ Amat) {
  int b = blockIdx.x;
  int wave = threadIdx.x >> 5;          // = m-tile
  int lane = threadIdx.x & 31;
  int row = lane & 15, hi = lane >> 4;
  const float* Ar = G + ((size_t)b * C_ + wave * 16 + row) * C_;
  for (int nt = 0; nt < 8; nt++) {
    const float* Br = We + (size_t)(nt * 16 + row) * C_;  // B col j = W_e row j
    v8f acc = {};
#pragma unroll
    for (int kt = 0; kt < 4; kt++) {
      int k0 = kt * 32;
      v16bf av, bv;
#pragma unroll
      for (int i = 0; i < 8; i++) {
        av[i]     = f2bf(Ar[k0 + hi * 8 + i]);
        av[i + 8] = f2bf(Ar[k0 + 16 + hi * 8 + i]);
        bv[i]     = f2bf(Br[k0 + hi * 8 + i]);
        bv[i + 8] = f2bf(Br[k0 + 16 + hi * 8 + i]);
      }
      acc = __builtin_amdgcn_wmma_f32_16x16x32_bf16(false, av, false, bv,
                                                    (short)0, acc, false, false);
    }
#pragma unroll
    for (int r = 0; r < 8; r++) {
      int M = r + (hi ? 8 : 0);
      Amat[((size_t)b * C_ + wave * 16 + M) * C_ + nt * 16 + (lane & 15)] = acc[r];
    }
  }
}

// ------------- dual softmax: A1 = softmax rows, Bm = softmax cols ----------
__global__ __launch_bounds__(256) void softmax_kernel(const float* __restrict__ Amat,
                                                      bf16* __restrict__ A1,
                                                      bf16* __restrict__ Bm) {
  int wave = threadIdx.x >> 5, lane = threadIdx.x & 31;
  int rowid = blockIdx.x * 8 + wave;          // 0 .. 2*B*C
  bool colmode = rowid >= B_ * C_;
  int rid = colmode ? rowid - B_ * C_ : rowid;
  int b = rid >> 7, r = rid & 127;
  float v[4];
#pragma unroll
  for (int k = 0; k < 4; k++) {
    int e = lane + 32 * k;
    v[k] = colmode ? Amat[((size_t)b * C_ + e) * C_ + r]
                   : Amat[((size_t)b * C_ + r) * C_ + e];
  }
  float mx = fmaxf(fmaxf(v[0], v[1]), fmaxf(v[2], v[3]));
#pragma unroll
  for (int off = 16; off >= 1; off >>= 1) mx = fmaxf(mx, __shfl_xor(mx, off, 32));
  float s = 0.f;
#pragma unroll
  for (int k = 0; k < 4; k++) { v[k] = __expf(v[k] - mx); s += v[k]; }
#pragma unroll
  for (int off = 16; off >= 1; off >>= 1) s += __shfl_xor(s, off, 32);
  float inv = 1.0f / s;
  bf16* out = colmode ? Bm : A1;
#pragma unroll
  for (int k = 0; k < 4; k++)
    out[((size_t)b * C_ + r) * C_ + lane + 32 * k] = f2bf(v[k] * inv);
}

// -------- GEMM3/4: y = att @ X + residual   (M=C tile, K=C, N big) ---------
// B-panel (4KB contiguous in [N][C] layout) staged to LDS with the CDNA5
// async-to-LDS path (ASYNCcnt), then consumed with ds_load_b128.
__global__ __launch_bounds__(256) void attgemm_kernel(
    const bf16* __restrict__ A1, const bf16* __restrict__ Bm,
    const bf16* __restrict__ qbT, const bf16* __restrict__ ebT,
    const float* __restrict__ exemplar, const float* __restrict__ query,
    bf16* __restrict__ ye, bf16* __restrict__ yq) {
  int branch = blockIdx.y;                    // 0: exemplar path, 1: query path
  const bf16* Aop = branch ? A1 : Bm;
  const bf16* Bop = branch ? ebT : qbT;
  const float* add = branch ? query : exemplar;
  bf16* out = branch ? yq : ye;
  int blk = blockIdx.x;                       // B*8*128
  int b  = blk >> 10;
  int mt = (blk >> 7) & 7;
  int ng = blk & 127;
  int wave = threadIdx.x >> 5, lane = threadIdx.x & 31;
  int nt = ng * 8 + wave;
  int row = lane & 15, hi = lane >> 4;

  __shared__ bf16 panel[8][16 * C_];          // 8 waves x 4KB
  // async-copy this wave's 4KB B panel into its private LDS region
  unsigned lbase = (unsigned)(size_t)&panel[wave][0];
  unsigned long long gbase =
      (unsigned long long)(size_t)(Bop + ((size_t)b * N_ + nt * 16) * C_);
#pragma unroll
  for (int j = 0; j < 8; j++) {
    unsigned loff = lbase + (unsigned)(j * 512 + lane * 16);
    unsigned long long ga = gbase + (unsigned long long)(j * 512 + lane * 16);
    asm volatile("global_load_async_to_lds_b128 %0, %1, off"
                 :: "v"(loff), "v"(ga) : "memory");
  }
  const bf16* Ar = Aop + ((size_t)b * C_ + mt * 16 + row) * C_;
  const bf16* Bl = &panel[wave][row * C_];    // B col j = row, K-contiguous
  asm volatile("s_wait_asynccnt 0x0" ::: "memory");

  v8f acc = {};
#pragma unroll
  for (int kt = 0; kt < 4; kt++) {
    int k0 = kt * 32;
    v8bf a0 = *(const v8bf*)(Ar + k0 + hi * 8);
    v8bf a1 = *(const v8bf*)(Ar + k0 + 16 + hi * 8);
    v8bf b0 = *(const v8bf*)(Bl + k0 + hi * 8);         // ds_load_b128
    v8bf b1 = *(const v8bf*)(Bl + k0 + 16 + hi * 8);
    v16bf av, bv;
#pragma unroll
    for (int i = 0; i < 8; i++) {
      av[i] = a0[i]; av[i + 8] = a1[i];
      bv[i] = b0[i]; bv[i + 8] = b1[i];
    }
    acc = __builtin_amdgcn_wmma_f32_16x16x32_bf16(false, av, false, bv,
                                                  (short)0, acc, false, false);
  }
#pragma unroll
  for (int r = 0; r < 8; r++) {
    int M = r + (hi ? 8 : 0);
    size_t idx = ((size_t)b * C_ + mt * 16 + M) * (size_t)N_ + nt * 16 + (lane & 15);
    out[idx] = f2bf(acc[r] + add[idx]);
  }
}

// ---------- depthwise 3x3 + BN + PReLU; writes transposed [B][N][C] --------
__global__ __launch_bounds__(256) void dwconv_kernel(
    const bf16* __restrict__ ye, const bf16* __restrict__ yq,
    const float* __restrict__ w9, const float* __restrict__ g,
    const float* __restrict__ be, const float* __restrict__ mn,
    const float* __restrict__ vr, const float* __restrict__ al,
    bf16* __restrict__ te, bf16* __restrict__ tq) {
  int branch = blockIdx.y;
  const bf16* y = branch ? yq : ye;
  bf16* t = branch ? tq : te;
  int gid = blockIdx.x * 256 + threadIdx.x;   // B*C*N threads
  int hw = gid & (N_ - 1);
  int c  = (gid >> 14) & 127;
  int b  = gid >> 21;
  int h = hw >> 7, w = hw & 127;
  const bf16* img = y + ((size_t)b * C_ + c) * N_;
  const float* wk = w9 + c * 9;
  float acc = 0.f;
#pragma unroll
  for (int dh = -1; dh <= 1; dh++) {
    int hh = h + dh;
    if ((unsigned)hh >= 128u) continue;
#pragma unroll
    for (int dw = -1; dw <= 1; dw++) {
      int ww = w + dw;
      if ((unsigned)ww >= 128u) continue;
      acc += bf2f(img[hh * 128 + ww]) * wk[(dh + 1) * 3 + (dw + 1)];
    }
  }
  float inv = g[c] * rsqrtf(vr[c] + 1e-5f);
  float v = acc * inv + (be[c] - mn[c] * inv);
  v = v >= 0.f ? v : al[c] * v;
  t[((size_t)b * N_ + hw) * C_ + c] = f2bf(v);
}

// --------------- pointwise 1x1 GEMM + BN + PReLU -> fp32 out ---------------
// B-panel fetched by the Tensor Data Mover: one tensor_load_to_lds per wave
// (1-D descriptor, data_size=2B, tile_dim0=2048 -> 4KB), S_WAIT_TENSORCNT.
__global__ __launch_bounds__(256) void pwconv_kernel(
    const bf16* __restrict__ pwb, const bf16* __restrict__ te,
    const bf16* __restrict__ tq,
    const float* __restrict__ g, const float* __restrict__ be,
    const float* __restrict__ mn, const float* __restrict__ vr,
    const float* __restrict__ al,
    float* __restrict__ out0, float* __restrict__ out1) {
  int branch = blockIdx.y;
  const bf16* t = branch ? tq : te;
  float* out = branch ? out1 : out0;
  int blk = blockIdx.x;                       // B*8*128
  int b  = blk >> 10;
  int mt = (blk >> 7) & 7;
  int ng = blk & 127;
  int wave = threadIdx.x >> 5, lane = threadIdx.x & 31;
  int nt = ng * 8 + wave;
  int row = lane & 15, hi = lane >> 4;

  __shared__ bf16 panel[8][16 * C_];          // 8 waves x 4KB
  // --- TDM descriptor (group0 128b + group1 256b), wave-uniform ---
  unsigned long long gaddr =
      (unsigned long long)(size_t)(t + ((size_t)b * N_ + nt * 16) * C_);
  u32x4 g0;
  g0[0] = 1u;                                  // count=1, user mode
  g0[1] = (unsigned)(size_t)&panel[wave][0];   // lds_addr
  g0[2] = (unsigned)(gaddr & 0xFFFFFFFFu);     // global_addr[31:0]
  g0[3] = ((unsigned)(gaddr >> 32) & 0x01FFFFFFu) | 0x80000000u; // addr|type=2
  u32x8 g1;
  g1[0] = 0x00010000u;                         // data_size=1 (2 bytes)
  g1[1] = 2048u << 16;                         // tensor_dim0 = 2048 (lo16)
  g1[2] = 1u << 16;                            // tensor_dim1 = 1
  g1[3] = 2048u << 16;                         // tile_dim0 = 2048
  g1[4] = 1u;                                  // tile_dim1 = 1, tile_dim2 = 0
  g1[5] = 2048u;                               // tensor_dim0_stride = 2048
  g1[6] = 0u;
  g1[7] = 0u;
  asm volatile("tensor_load_to_lds %0, %1" :: "s"(g0), "s"(g1) : "memory");

  const bf16* Ar = pwb + (size_t)(mt * 16 + row) * C_;
  const bf16* Bl = &panel[wave][row * C_];
  __builtin_amdgcn_s_wait_tensorcnt(0);

  v8f acc = {};
#pragma unroll
  for (int kt = 0; kt < 4; kt++) {
    int k0 = kt * 32;
    v8bf a0 = *(const v8bf*)(Ar + k0 + hi * 8);
    v8bf a1 = *(const v8bf*)(Ar + k0 + 16 + hi * 8);
    v8bf b0 = *(const v8bf*)(Bl + k0 + hi * 8);         // ds_load_b128
    v8bf b1 = *(const v8bf*)(Bl + k0 + 16 + hi * 8);
    v16bf av, bv;
#pragma unroll
    for (int i = 0; i < 8; i++) {
      av[i] = a0[i]; av[i + 8] = a1[i];
      bv[i] = b0[i]; bv[i + 8] = b1[i];
    }
    acc = __builtin_amdgcn_wmma_f32_16x16x32_bf16(false, av, false, bv,
                                                  (short)0, acc, false, false);
  }
#pragma unroll
  for (int r = 0; r < 8; r++) {
    int o = mt * 16 + r + (hi ? 8 : 0);
    float inv = g[o] * rsqrtf(vr[o] + 1e-5f);
    float v = acc[r] * inv + (be[o] - mn[o] * inv);
    v = v >= 0.f ? v : al[o] * v;
    out[((size_t)b * C_ + o) * (size_t)N_ + nt * 16 + (lane & 15)] = v;
  }
}

// ---------------------------------------------------------------------------
extern "C" void kernel_launch(void* const* d_in, const int* in_sizes, int n_in,
                              void* d_out, int out_size, void* d_ws, size_t ws_size,
                              hipStream_t stream) {
  const float* exemplar = (const float*)d_in[0];
  const float* query    = (const float*)d_in[1];
  const float* We       = (const float*)d_in[2];
  const float* dw_w     = (const float*)d_in[3];
  const float* dw_gamma = (const float*)d_in[4];
  const float* dw_beta  = (const float*)d_in[5];
  const float* dw_mean  = (const float*)d_in[6];
  const float* dw_var   = (const float*)d_in[7];
  const float* dw_alpha = (const float*)d_in[8];
  const float* pw_w     = (const float*)d_in[9];
  const float* pw_gamma = (const float*)d_in[10];
  const float* pw_beta  = (const float*)d_in[11];
  const float* pw_mean  = (const float*)d_in[12];
  const float* pw_var   = (const float*)d_in[13];
  const float* pw_alpha = (const float*)d_in[14];
  (void)in_sizes; (void)n_in; (void)out_size; (void)ws_size;

  // workspace layout (bf16 unless noted); peak ~406 MB
  bf16* qb  = (bf16*)d_ws;          // [B][C][N]
  bf16* eb  = qb  + BCN;
  bf16* qbT = eb  + BCN;            // [B][N][C]
  bf16* ebT = qbT + BCN;
  bf16* ye  = ebT + BCN;            // conv inputs
  bf16* yq  = ye  + BCN;
  bf16* te  = qb;                   // reuse: qb/eb dead after gemm1
  bf16* tq  = eb;
  float* G    = (float*)(yq + BCN); // [B][C][C] fp32
  float* Amat = G + BCC;
  bf16* A1  = (bf16*)(Amat + BCC);
  bf16* Bm  = A1 + BCC;
  bf16* pwb = Bm + BCC;             // pw_w in bf16

  cvt_kernel<<<(int)(BCN / 4 / 256), 256, 0, stream>>>(query, qb, (int)(BCN / 4));
  cvt_kernel<<<(int)(BCN / 4 / 256), 256, 0, stream>>>(exemplar, eb, (int)(BCN / 4));
  cvt_kernel<<<16, 256, 0, stream>>>(pw_w, pwb, C_ * C_ / 4);
  transpose_kernel<<<B_ * 4 * 512, 256, 0, stream>>>(query, qbT);
  transpose_kernel<<<B_ * 4 * 512, 256, 0, stream>>>(exemplar, ebT);

  gemm1_kernel<<<B_ * 8 * 8, 256, 0, stream>>>(qb, eb, G);
  gemm2_kernel<<<B_, 256, 0, stream>>>(G, We, Amat);
  softmax_kernel<<<2 * B_ * C_ / 8, 256, 0, stream>>>(Amat, A1, Bm);

  attgemm_kernel<<<dim3(B_ * 8 * 128, 2), 256, 0, stream>>>(
      A1, Bm, qbT, ebT, exemplar, query, ye, yq);

  dwconv_kernel<<<dim3((int)(BCN / 256), 2), 256, 0, stream>>>(
      ye, yq, dw_w, dw_gamma, dw_beta, dw_mean, dw_var, dw_alpha, te, tq);

  pwconv_kernel<<<dim3(B_ * 8 * 128, 2), 256, 0, stream>>>(
      pwb, te, tq, pw_gamma, pw_beta, pw_mean, pw_var, pw_alpha,
      (float*)d_out, (float*)d_out + BCN);
}
